// GroupedQueryAttention_70265664963162
// MI455X (gfx1250) — matrix operand order repaired and checked by reading, not compiled, not run on hardware
//
#include <hip/hip_runtime.h>
#include <hip/hip_bf16.h>

typedef __attribute__((ext_vector_type(16))) _Float16 v16h;
typedef __attribute__((ext_vector_type(8)))  float    v8f;
typedef __attribute__((ext_vector_type(4)))  int      v4i;

#define B_    2
#define S_    2048
#define D_    1024
#define H_    16
#define HKV_  4
#define DK_   64
#define RATIO_ 4
#define DKV_  (HKV_ * DK_)   // 256

#if defined(__gfx1250__) && __has_builtin(__builtin_amdgcn_global_load_async_to_lds_b128)
#define USE_ASYNC_LDS 1
__device__ __forceinline__ void async_b128(const void* g, const void* l) {
    // toolchain-probed signature: (v4i* global_src, v4i* lds_dst, imm offset, imm cpol)
    __builtin_amdgcn_global_load_async_to_lds_b128((v4i*)g, (v4i*)l, 0, 0);
}
__device__ __forceinline__ void async_wait0() {
#if __has_builtin(__builtin_amdgcn_s_wait_asynccnt)
    __builtin_amdgcn_s_wait_asynccnt(0);
#else
    asm volatile("s_wait_asynccnt 0x0" ::: "memory");
#endif
}
#else
#define USE_ASYNC_LDS 0
#endif

__device__ __forceinline__ v8f zero8() {
    v8f z = {0.f, 0.f, 0.f, 0.f, 0.f, 0.f, 0.f, 0.f};
    return z;
}

// ---------------------------------------------------------------------------
// WMMA GEMM: C[M,N] = A[M,K] * B[K,N] + bias[N]
// Block: 128 threads = 4 waves. Tile 64x64, K-step 32.
// Each wave computes a 32x32 subtile as 2x2 v_wmma_f32_16x16x32_f16.
// A staged [m][k]; B staged TRANSPOSED [n][k] so both WMMA fragment loads
// are contiguous 16-byte LDS reads (ds_load_b128).
// ---------------------------------------------------------------------------
template <typename TA, typename TB, typename TC>
__global__ __launch_bounds__(128)
void gemm_wmma_kernel(const TA* __restrict__ A, const TB* __restrict__ Bm,
                      const float* __restrict__ bias, TC* __restrict__ C,
                      int M, int N, int K)
{
    __shared__ _Float16 As[64][40];   // [m][k]  64x32 (+8 pad)
    __shared__ _Float16 Bs[64][40];   // [n][k]  64x32 (+8 pad), transposed

    const int tid  = threadIdx.x;
    const int lane = tid & 31;
    const int wave = tid >> 5;
    const int half = lane >> 4;       // wave32 half-wave select
    const int l16  = lane & 15;

    const int bm = blockIdx.x * 64;
    const int bn = blockIdx.y * 64;
    const int wm = (wave >> 1) * 32;
    const int wn = (wave & 1) * 32;

    v8f acc[2][2];
    acc[0][0] = zero8(); acc[0][1] = zero8();
    acc[1][0] = zero8(); acc[1][1] = zero8();

    for (int k0 = 0; k0 < K; k0 += 32) {
        __syncthreads();
        // Stage A tile (64x32) [m][k], f32/f16 -> f16
        for (int i = tid; i < 64 * 32; i += 128) {
            const int r = i >> 5, c = i & 31;
            As[r][c] = (_Float16)A[(size_t)(bm + r) * K + (k0 + c)];
        }
        // Stage B tile (32x64) transposed into [n][k]
        for (int i = tid; i < 32 * 64; i += 128) {
            const int r = i >> 6, c = i & 63;      // r = k, c = n
            Bs[c][r] = (_Float16)Bm[(size_t)(k0 + r) * N + (bn + c)];
        }
        // Prefetch next K tiles toward cache while this tile computes
        if (k0 + 32 < K) {
            __builtin_prefetch(&A[(size_t)(bm + (tid & 63)) * K + (k0 + 32)], 0, 3);
            __builtin_prefetch(&Bm[(size_t)(k0 + 32 + (tid & 31)) * N + bn], 0, 3);
        }
        __syncthreads();

        #pragma unroll
        for (int mi = 0; mi < 2; ++mi) {
            // A fragment 16x32 f16: lanes 0-15 row M=lane K=0..15;
            // lanes 16-31 same rows, K=16..31
            v16h af;
            const int arow = wm + mi * 16 + l16;
            const int koff = half * 16;
            #pragma unroll
            for (int j = 0; j < 8; ++j) {
                af[j]     = As[arow][koff + j];
                af[j + 8] = As[arow][koff + 8 + j];
            }
            #pragma unroll
            for (int ni = 0; ni < 2; ++ni) {
                // B fragment 32x16 f16: lane column N=lane%16,
                // lanes 0-15 K=0..15, lanes 16-31 K=16..31 (contiguous in Bs[n][k])
                v16h bf;
                const int brow = wn + ni * 16 + l16;
                const int kb   = half * 16;
                #pragma unroll
                for (int j = 0; j < 16; ++j) bf[j] = Bs[brow][kb + j];
                acc[mi][ni] = __builtin_amdgcn_wmma_f32_16x16x32_f16(
                    false, af, false, bf, (short)0, acc[mi][ni], false, false);
            }
        }
    }

    // C/D 16x16 f32 layout: lane holds column N=lane%16, rows r + 8*(lane/16)
    #pragma unroll
    for (int mi = 0; mi < 2; ++mi) {
        #pragma unroll
        for (int ni = 0; ni < 2; ++ni) {
            const int col = bn + wn + ni * 16 + l16;
            const float bv = bias[col];
            #pragma unroll
            for (int r = 0; r < 8; ++r) {
                const int row = bm + wm + mi * 16 + r + half * 8;
                C[(size_t)row * N + col] = (TC)(acc[mi][ni][r] + bv);
            }
        }
    }
}

// ---------------------------------------------------------------------------
// Causal grouped-query flash attention.
// One block = one (batch, head, 64-row q tile). 128 threads = 4 waves;
// wave w owns q rows [w*16, w*16+16). Online softmax over causal K/V tiles.
// Q/K tiles staged via the CDNA5 async global->LDS engine when available
// (byte-exact copies, ASYNCcnt-tracked); V staged TRANSPOSED [dk][key] via
// the VGPR path (needs transpose) so PV B-fragments are contiguous.
// ---------------------------------------------------------------------------
__global__ __launch_bounds__(128)
void gqa_flash_kernel(const _Float16* __restrict__ Qh,
                      const _Float16* __restrict__ Kh,
                      const _Float16* __restrict__ Vh,
                      _Float16* __restrict__ Out)
{
    __shared__ _Float16 Qs[64][72];    // [q][dk]
    __shared__ _Float16 Ks[64][72];    // [key][dk]
    __shared__ _Float16 Vst[64][72];   // [dk][key]  (transposed)
    __shared__ _Float16 Ps[64][72];    // [q][key]

    const int tid  = threadIdx.x;
    const int lane = tid & 31;
    const int wave = tid >> 5;
    const int half = lane >> 4;
    const int l16  = lane & 15;

    const int nqt = S_ / 64;
    int idx = blockIdx.x;
    const int qt = idx % nqt; idx /= nqt;
    const int h  = idx % H_;  idx /= H_;
    const int b  = idx;
    const int kvh = h / RATIO_;
    const int q0  = qt * 64;

    const _Float16* Qp = Qh + (size_t)b * S_ * D_   + (size_t)h   * DK_;
    const _Float16* Kp = Kh + (size_t)b * S_ * DKV_ + (size_t)kvh * DK_;
    const _Float16* Vp = Vh + (size_t)b * S_ * DKV_ + (size_t)kvh * DK_;

    // Load Q tile (64 rows x 64 dk)
#if USE_ASYNC_LDS
    #pragma unroll
    for (int p = 0; p < 4; ++p) {                 // 512 x 16B chunks total
        const int ci = p * 128 + tid;             // chunk index
        const int r = ci >> 3;                    // 8 chunks per 128B row
        const int c = (ci & 7) * 8;               // element offset (halves)
        async_b128(&Qp[(size_t)(q0 + r) * D_ + c], &Qs[r][c]);
    }
#else
    for (int i = tid; i < 64 * 64; i += 128) {
        const int r = i >> 6, c = i & 63;
        Qs[r][c] = Qp[(size_t)(q0 + r) * D_ + c];
    }
#endif

    float m_i[8], l_i[8];
    v8f acc_o[4];
    #pragma unroll
    for (int r = 0; r < 8; ++r) { m_i[r] = -1e30f; l_i[r] = 0.f; }
    #pragma unroll
    for (int ni = 0; ni < 4; ++ni) acc_o[ni] = zero8();

    const float scale = 0.125f;  // 1/sqrt(64)

    for (int kt = 0; kt <= qt; ++kt) {   // causal: tiles at/below the diagonal
        __syncthreads();
#if USE_ASYNC_LDS
        #pragma unroll
        for (int p = 0; p < 4; ++p) {
            const int ci = p * 128 + tid;
            const int r = ci >> 3;
            const int c = (ci & 7) * 8;
            async_b128(&Kp[(size_t)(kt * 64 + r) * DKV_ + c], &Ks[r][c]);
        }
        for (int i = tid; i < 64 * 64; i += 128) {
            const int r = i >> 6, c = i & 63;      // r = key, c = dk
            Vst[c][r] = Vp[(size_t)(kt * 64 + r) * DKV_ + c];
        }
        async_wait0();
#else
        for (int i = tid; i < 64 * 64; i += 128) {
            const int r = i >> 6, c = i & 63;      // r = key, c = dk
            Ks[r][c]  = Kp[(size_t)(kt * 64 + r) * DKV_ + c];
            Vst[c][r] = Vp[(size_t)(kt * 64 + r) * DKV_ + c];
        }
#endif
        // Prefetch next K/V tile
        if (kt < qt) {
            __builtin_prefetch(&Kp[(size_t)((kt + 1) * 64 + (tid & 63)) * DKV_], 0, 3);
            __builtin_prefetch(&Vp[(size_t)((kt + 1) * 64 + (tid & 63)) * DKV_], 0, 3);
        }
        __syncthreads();

        // ---- scores: S = Q * K^T (wave's 16 rows x 64 keys) ----
        v8f accs[4];
        #pragma unroll
        for (int ni = 0; ni < 4; ++ni) accs[ni] = zero8();

        #pragma unroll
        for (int kk = 0; kk < 64; kk += 32) {
            v16h af;
            const int arow = wave * 16 + l16;
            const int koff = kk + half * 16;
            #pragma unroll
            for (int j = 0; j < 8; ++j) {
                af[j]     = Qs[arow][koff + j];
                af[j + 8] = Qs[arow][koff + 8 + j];
            }
            #pragma unroll
            for (int ni = 0; ni < 4; ++ni) {
                v16h bf;                    // B = K^T: contiguous rows of Ks
                const int key = ni * 16 + l16;
                const int kb  = kk + half * 16;
                #pragma unroll
                for (int j = 0; j < 16; ++j) bf[j] = Ks[key][kb + j];
                accs[ni] = __builtin_amdgcn_wmma_f32_16x16x32_f16(
                    false, af, false, bf, (short)0, accs[ni], false, false);
            }
        }

        // ---- causal mask + online softmax (row lives in one half-wave) ----
        const int gr_base = q0 + wave * 16 + half * 8;
        #pragma unroll
        for (int r = 0; r < 8; ++r) {
            const int grow = gr_base + r;
            float rowm = -1e30f;
            #pragma unroll
            for (int ni = 0; ni < 4; ++ni) {
                const int gcol = kt * 64 + ni * 16 + l16;
                float s = accs[ni][r] * scale;
                s = (gcol <= grow) ? s : -1e30f;
                accs[ni][r] = s;
                rowm = fmaxf(rowm, s);
            }
            #pragma unroll
            for (int o = 8; o >= 1; o >>= 1)
                rowm = fmaxf(rowm, __shfl_xor(rowm, o, 16));
            const float mnew = fmaxf(m_i[r], rowm);
            const float corr = __expf(m_i[r] - mnew);
            m_i[r] = mnew;
            float rs = 0.f;
            #pragma unroll
            for (int ni = 0; ni < 4; ++ni) {
                const float p = __expf(accs[ni][r] - mnew);
                accs[ni][r] = p;
                rs += p;
            }
            #pragma unroll
            for (int o = 8; o >= 1; o >>= 1)
                rs += __shfl_xor(rs, o, 16);
            l_i[r] = l_i[r] * corr + rs;
            #pragma unroll
            for (int ni = 0; ni < 4; ++ni) acc_o[ni][r] *= corr;
        }

        // ---- P -> LDS (f16) so it can be re-read as WMMA A fragments ----
        #pragma unroll
        for (int ni = 0; ni < 4; ++ni)
            #pragma unroll
            for (int r = 0; r < 8; ++r)
                Ps[wave * 16 + r + half * 8][ni * 16 + l16] =
                    (_Float16)accs[ni][r];
        __syncthreads();

        // ---- O += P * V ----
        #pragma unroll
        for (int kk = 0; kk < 64; kk += 32) {
            v16h af;
            const int arow = wave * 16 + l16;
            const int koff = kk + half * 16;
            #pragma unroll
            for (int j = 0; j < 8; ++j) {
                af[j]     = Ps[arow][koff + j];
                af[j + 8] = Ps[arow][koff + 8 + j];
            }
            #pragma unroll
            for (int ni = 0; ni < 4; ++ni) {
                v16h bf;                    // contiguous rows of Vst[dk][key]
                const int dkr = ni * 16 + l16;
                const int kb  = kk + half * 16;
                #pragma unroll
                for (int j = 0; j < 16; ++j) bf[j] = Vst[dkr][kb + j];
                acc_o[ni] = __builtin_amdgcn_wmma_f32_16x16x32_f16(
                    false, af, false, bf, (short)0, acc_o[ni], false, false);
            }
        }
    }

    // ---- normalize and write attention output [B,S,H*DK] as f16 ----
    #pragma unroll
    for (int ni = 0; ni < 4; ++ni) {
        #pragma unroll
        for (int r = 0; r < 8; ++r) {
            const int grow = q0 + wave * 16 + half * 8 + r;
            const float o = acc_o[ni][r] / l_i[r];
            Out[(size_t)(b * S_ + grow) * D_ + h * DK_ + ni * 16 + l16] =
                (_Float16)o;
        }
    }
}

// ---------------------------------------------------------------------------
extern "C" void kernel_launch(void* const* d_in, const int* in_sizes, int n_in,
                              void* d_out, int out_size, void* d_ws, size_t ws_size,
                              hipStream_t stream)
{
    const float* q  = (const float*)d_in[0];
    const float* k  = (const float*)d_in[1];
    const float* v  = (const float*)d_in[2];
    // d_in[3] = mask: exactly causal, computed analytically in-kernel
    const float* Wq = (const float*)d_in[4];
    const float* bq = (const float*)d_in[5];
    const float* Wk = (const float*)d_in[6];
    const float* bk = (const float*)d_in[7];
    const float* Wv = (const float*)d_in[8];
    const float* bv = (const float*)d_in[9];
    const float* Wo = (const float*)d_in[10];
    const float* bo = (const float*)d_in[11];
    float* out = (float*)d_out;

    const int M = B_ * S_;                       // 4096
    _Float16* qh16 = (_Float16*)d_ws;            // 4096x1024
    _Float16* kh16 = qh16 + (size_t)M * D_;      // 4096x256
    _Float16* vh16 = kh16 + (size_t)M * DKV_;    // 4096x256
    _Float16* ao16 = vh16 + (size_t)M * DKV_;    // 4096x1024
    (void)in_sizes; (void)n_in; (void)out_size; (void)ws_size;

    dim3 blk(128);

    // Projections (f32 in -> f16 intermediate, resident in L2)
    gemm_wmma_kernel<float, float, _Float16>
        <<<dim3(M / 64, D_ / 64), blk, 0, stream>>>(q, Wq, bq, qh16, M, D_, D_);
    gemm_wmma_kernel<float, float, _Float16>
        <<<dim3(M / 64, DKV_ / 64), blk, 0, stream>>>(k, Wk, bk, kh16, M, DKV_, D_);
    gemm_wmma_kernel<float, float, _Float16>
        <<<dim3(M / 64, DKV_ / 64), blk, 0, stream>>>(v, Wv, bv, vh16, M, DKV_, D_);

    // Causal GQA flash attention
    gqa_flash_kernel<<<dim3(B_ * H_ * (S_ / 64)), blk, 0, stream>>>(
        qh16, kh16, vh16, ao16);

    // Output projection (f16 A, f32 W/out)
    gemm_wmma_kernel<_Float16, float, float>
        <<<dim3(M / 64, D_ / 64), blk, 0, stream>>>(ao16, Wo, bo, out, M, D_, D_);
}
